// SSMLayer_43954695307847
// MI455X (gfx1250) — matrix operand order repaired and checked by reading, not compiled, not run on hardware
//
#include <hip/hip_runtime.h>
#include <hip/hip_bf16.h>
#include <math.h>

// ---------------- problem constants (from reference) ----------------
static constexpr int kB   = 4;
static constexpr int kT   = 4096;
static constexpr int kDm  = 1024;
static constexpr int kDn  = 2048;          // 2 * kDm
static constexpr int kM   = kB * kT;       // 16384 tokens
static constexpr int kDtRank = 64;
static constexpr int kNState = 16;
static constexpr int kXbcW   = kDtRank + 2 * kNState; // 96
static constexpr int kDConv  = 4;

// ---------------- CDNA5 WMMA / TDM types ----------------
typedef __bf16 bf16_t;
typedef __attribute__((ext_vector_type(16))) __bf16 v16bf;
typedef __attribute__((ext_vector_type(8)))  __bf16 v8bf;
typedef __attribute__((ext_vector_type(8)))  float  v8f;
typedef __attribute__((ext_vector_type(4)))  unsigned int v4u;
typedef __attribute__((ext_vector_type(8)))  int v8i;
typedef __attribute__((ext_vector_type(4)))  int v4i;

#if __has_builtin(__builtin_amdgcn_tensor_load_to_lds)
#define HAVE_TDM 1
#endif

static __device__ __forceinline__ void wait_tensorcnt0() {
#if __has_builtin(__builtin_amdgcn_s_wait_tensorcnt)
  __builtin_amdgcn_s_wait_tensorcnt(0);
#else
  asm volatile("s_wait_tensorcnt 0x0" ::: "memory");
#endif
}

// ---------------- bf16 helpers (RNE, bit-exact) -----
static __device__ __forceinline__ unsigned short f2bf(float f) {
  union { float f; unsigned u; } v; v.f = f;
  unsigned u = v.u;
  unsigned r = (u + 0x7FFFu + ((u >> 16) & 1u)) >> 16;
  return (unsigned short)r;
}
static __device__ __forceinline__ float bf2f(unsigned short h) {
  union { unsigned u; float f; } v; v.u = ((unsigned)h) << 16;
  return v.f;
}

// ---------------- fragment loaders (layouts per CDNA5 ISA 7.12.2) ----------
// A: 16x32 bf16. lane L -> row m = L%16, half = L/16;
// elems 0..7 : k0+8*half+0..7 ; elems 8..15 : k0+16+8*half+0..7
static __device__ __forceinline__ v16bf frag_from(const bf16_t* p) {
  v8bf lo = *(const v8bf*)p;
  v8bf hi = *(const v8bf*)(p + 16);
  v16bf r;
#pragma unroll
  for (int i = 0; i < 8; ++i) { r[i] = lo[i]; r[i + 8] = hi[i]; }
  return r;
}
static __device__ __forceinline__ v16bf load_a_frag_g(const bf16_t* __restrict__ A,
                                                      int lda, int row0, int k0, int lane) {
  return frag_from(A + (size_t)(row0 + (lane & 15)) * lda + k0 + 8 * (lane >> 4));
}
// B for C = A @ W^T, W row-major [N][K]: W row n, same k order as A fragment.
static __device__ __forceinline__ v16bf load_b_frag(const bf16_t* __restrict__ W,
                                                    int ldw, int n0, int k0, int lane) {
  const bf16_t* p = W + (size_t)(n0 + (lane & 15)) * ldw + k0 + 8 * (lane >> 4);
  return frag_from(p);
}

// ---------------- epilogue helper ----------------
// EPI: 0 = store bf16 | 1 = store f32 | 2 = dt epilogue (softplus, bf16)
//      3 = +residual, store f32
template <int EPI>
static __device__ __forceinline__ void store_tile(const v8f& acc, int row0, int n0,
                                                  int N, int lane,
                                                  const float* __restrict__ extra,
                                                  float* __restrict__ outF,
                                                  unsigned short* __restrict__ outB) {
  const int n    = n0 + (lane & 15);
  const int hrow = (lane >> 4) * 8;
#pragma unroll
  for (int v = 0; v < 8; ++v) {
    const size_t m = (size_t)(row0 + hrow + v);
    float val = acc[v];
    if (EPI == 0) {
      outB[m * N + n] = f2bf(val);
    } else if (EPI == 1) {
      outF[m * N + n] = val;
    } else if (EPI == 2) {
      val += extra[n];                                   // + b_dt
      float sp = (val > 20.f) ? val : logf(1.f + __expf(val));
      float dt = fminf(fmaxf(sp + 1e-3f, 1e-3f), 0.1f);  // clip [1e-3, 0.1]
      outB[m * N + n] = f2bf(dt);
    } else {                                             // EPI == 3
      val += extra[m * N + n];                           // + residual
      outF[m * N + n] = val;
    }
  }
}

// ---------------- TDM: load a 16 x KC bf16 panel (row stride K) into LDS ---
// LDS layout: row stride (KC+8) bf16 (16B pad per row -> bank-conflict free).
// D# per CDNA5 ISA ch.8: group0 {count,lds,global,type=2},
// group1 {data_size=2B, pad_enable, dims, tile, stride0}.
// This toolchain's builtin is the 6-arg form:
//   (uint32x4 g0, int32x8 g1, int32x4 g2, int32x4 g3, int32x8 gx, i32 cpol)
template <int KC>
static __device__ __forceinline__ void tdm_load_a_panel(const bf16_t* gsrc,
                                                        unsigned lds_off, int K_len) {
#ifdef HAVE_TDM
  constexpr unsigned padIntCode = (KC == 128) ? 5u : 4u;  // 2^(c+1) dwords = KC/2
  const unsigned long long ga = (unsigned long long)(size_t)gsrc;
  v4u g0;
  g0[0] = 1u;                                        // count=1 (valid user D#)
  g0[1] = lds_off;                                   // lds_addr (bytes)
  g0[2] = (unsigned)(ga & 0xFFFFFFFFu);              // global_addr[31:0]
  g0[3] = (unsigned)((ga >> 32) & 0x01FFFFFFu) | (2u << 30);  // [120:64] addr, type=2
  const unsigned td0 = (unsigned)K_len;              // tensor_dim0 (elements)
  v8i g1;
  g1[0] = (int)((1u << 16)                           // data_size = 1 -> 2 bytes
              | (1u << 20)                           // pad_enable
              | (padIntCode << 22)                   // pad_interval
              | (3u << 25));                         // pad_amount = 4 dwords (8 bf16)
  g1[1] = (int)((td0 & 0xFFFFu) << 16);              // abar=0 | tensor_dim0[15:0]
  g1[2] = (int)(((td0 >> 16) & 0xFFFFu) | (16u << 16)); // td0[31:16] | tensor_dim1=16
  g1[3] = (int)((unsigned)KC << 16);                 // td1[31:16]=0 | tile_dim0=KC
  g1[4] = 16;                                        // tile_dim1=16, tile_dim2=0
  g1[5] = (int)td0;                                  // tensor_dim0_stride = K
  g1[6] = 0;                                         // stride0[47:32]=0, stride1=0
  g1[7] = 0;
  v4i g2 = {0, 0, 0, 0};                             // 2-D tensor: group2 unused
  v4i g3 = {0, 0, 0, 0};                             // 2-D tensor: group3 unused
  v8i gx = {0, 0, 0, 0, 0, 0, 0, 0};                 // extra operand (unused dims)
  __builtin_amdgcn_tensor_load_to_lds(g0, g1, g2, g3, gx, 0);
#else
  (void)gsrc; (void)lds_off; (void)K_len;
#endif
}

// ---------------- GEMM with LDS-staged A panel ----------------
// Block = 256 thr = 8 waves; block tile 16(M) x 512(N); wave tile 16 x 64.
// Requires: M%16==0, N%512==0, K%KC==0.
template <int EPI, int KC>
__global__ void __launch_bounds__(256)
gemm_lds_wmma(const bf16_t* __restrict__ A, const bf16_t* __restrict__ W,
              int M, int N, int K,
              const float* __restrict__ extra,
              float* __restrict__ outF, unsigned short* __restrict__ outB) {
  constexpr int LROW = KC + 8;                       // padded LDS row (bf16 elems)
  __shared__ __align__(16) bf16_t sm[16 * LROW];

  const int lane  = threadIdx.x & 31;
  const int wv    = threadIdx.x >> 5;                // 0..7
  const int tnb   = blockIdx.x % (N >> 9);
  const int tm    = blockIdx.x / (N >> 9);
  const int row0  = tm << 4;
  const int ncol0 = (tnb << 9) + (wv << 6);

  v8f acc[4] = {v8f{}, v8f{}, v8f{}, v8f{}};

  for (int kc = 0; kc < K; kc += KC) {
    __syncthreads();                                 // previous panel consumed
#ifdef HAVE_TDM
    if (threadIdx.x < 32) {                          // wave 0 drives the TDM
      tdm_load_a_panel<KC>(A + (size_t)row0 * K + kc,
                           (unsigned)(unsigned long long)(&sm[0]), K);
      wait_tensorcnt0();
    }
#else
    // cooperative, coalesced copy: 16 rows x KC bf16, 8-elem chunks
    for (int c = threadIdx.x; c < 16 * (KC / 8); c += 256) {
      const int r = c / (KC / 8), cc = (c % (KC / 8)) * 8;
      *(v8bf*)&sm[r * LROW + cc] =
          *(const v8bf*)&A[(size_t)(row0 + r) * K + kc + cc];
    }
#endif
    __syncthreads();                                 // panel visible

#pragma unroll
    for (int k0 = 0; k0 < KC; k0 += 32) {
      const v16bf a = frag_from(&sm[(lane & 15) * LROW + k0 + 8 * (lane >> 4)]);
#pragma unroll
      for (int s = 0; s < 4; ++s) {
        const v16bf b = load_b_frag(W, K, ncol0 + s * 16, kc + k0, lane);
        acc[s] = __builtin_amdgcn_wmma_f32_16x16x32_bf16(
            false, a, false, b, (short)0, acc[s], false, false);
      }
    }
  }
#pragma unroll
  for (int s = 0; s < 4; ++s)
    store_tile<EPI>(acc[s], row0, ncol0 + s * 16, N, lane, extra, outF, outB);
}

// ---------------- plain GEMM (small N; no LDS). N == 16*NSUB ----------------
template <int EPI, int NSUB>
__global__ void __launch_bounds__(256)
gemm_plain_wmma(const bf16_t* __restrict__ A, const bf16_t* __restrict__ W,
                int M, int N, int K,
                const float* __restrict__ extra,
                float* __restrict__ outF, unsigned short* __restrict__ outB) {
  const int lane = threadIdx.x & 31;
  const int wave = blockIdx.x * (blockDim.x >> 5) + (threadIdx.x >> 5);
  const int row0 = wave << 4;                        // one wave per 16-row strip

  v8f acc[NSUB];
#pragma unroll
  for (int s = 0; s < NSUB; ++s) acc[s] = v8f{};

  for (int k0 = 0; k0 < K; k0 += 32) {
    if (k0 + 32 < K) {
      const bf16_t* pf = A + (size_t)(row0 + (lane & 15)) * K + (k0 + 32);
      __builtin_prefetch((const void*)pf, 0, 0);
    }
    const v16bf a = load_a_frag_g(A, K, row0, k0, lane);
#pragma unroll
    for (int s = 0; s < NSUB; ++s) {
      const v16bf b = load_b_frag(W, K, s * 16, k0, lane);
      acc[s] = __builtin_amdgcn_wmma_f32_16x16x32_bf16(
          false, a, false, b, (short)0, acc[s], false, false);
    }
  }
#pragma unroll
  for (int s = 0; s < NSUB; ++s)
    store_tile<EPI>(acc[s], row0, s * 16, N, lane, extra, outF, outB);
}

// ---------------- LayerNorm (f32 stats, bf16 out) ----------------
__global__ void __launch_bounds__(256)
ln_kernel(const float* __restrict__ x, const float* __restrict__ w,
          const float* __restrict__ b, unsigned short* __restrict__ xn) {
  __shared__ float s1[256], s2[256];
  const int tid = threadIdx.x;
  const size_t m = blockIdx.x;
  const float* row = x + m * kDm;
  float a = 0.f, q = 0.f;
  for (int i = tid; i < kDm; i += 256) { float v = row[i]; a += v; q += v * v; }
  s1[tid] = a; s2[tid] = q; __syncthreads();
  for (int off = 128; off > 0; off >>= 1) {
    if (tid < off) { s1[tid] += s1[tid + off]; s2[tid] += s2[tid + off]; }
    __syncthreads();
  }
  const float mu   = s1[0] * (1.f / kDm);
  const float var  = s2[0] * (1.f / kDm) - mu * mu;
  const float rstd = rsqrtf(var + 1e-5f);
  for (int i = tid; i < kDm; i += 256)
    xn[m * kDm + i] = f2bf((row[i] - mu) * rstd * w[i] + b[i]);
}

// ---------------- f32 -> bf16 convert ----------------
__global__ void __launch_bounds__(256)
cvt_bf16_kernel(const float* __restrict__ in, unsigned short* __restrict__ out, int n) {
  int i = blockIdx.x * blockDim.x + threadIdx.x;
  if (i < n) out[i] = f2bf(in[i]);
}

// ---------------- causal depthwise conv (k=4) + SiLU ----------------
__global__ void __launch_bounds__(256)
conv_silu_kernel(const unsigned short* __restrict__ xz, const float* __restrict__ cw,
                 const float* __restrict__ cb, unsigned short* __restrict__ u) {
  const int idx = blockIdx.x * blockDim.x + threadIdx.x;
  if (idx >= kM * kDn) return;
  const int d  = idx % kDn;
  const int mt = idx / kDn;          // b*T + t
  const int t  = mt % kT;
  float acc = cb[d];
#pragma unroll
  for (int i = 0; i < kDConv; ++i) {
    const int tt = t - (kDConv - 1) + i;
    if (tt >= 0)
      acc += bf2f(xz[(size_t)(mt - t + tt) * (2 * kDn) + d]) * cw[d * kDConv + i];
  }
  const float s = acc / (1.f + __expf(-acc));
  u[(size_t)mt * kDn + d] = f2bf(s);
}

// ---------------- slice xbc[:, :64] -> bf16 ----------------
__global__ void __launch_bounds__(256)
slice_dtlow_kernel(const float* __restrict__ xbc, unsigned short* __restrict__ dtl) {
  const int i = blockIdx.x * blockDim.x + threadIdx.x;
  if (i >= kM * kDtRank) return;
  const int m = i >> 6, c = i & 63;
  dtl[i] = f2bf(xbc[(size_t)m * kXbcW + c]);
}

// ---------------- selective scan: one channel (b,d) per lane -------------
__global__ void __launch_bounds__(256)
scan_kernel(const unsigned short* __restrict__ dtb, const unsigned short* __restrict__ ub,
            const float* __restrict__ xbc, const unsigned short* __restrict__ xzb,
            const float* __restrict__ logA, const float* __restrict__ Dp,
            unsigned short* __restrict__ yb) {
  const int c = blockIdx.x * blockDim.x + threadIdx.x;
  if (c >= kB * kDn) return;
  const int d = c % kDn;
  const int b = c / kDn;

  float Ad[kNState];
#pragma unroll
  for (int n = 0; n < kNState; ++n) Ad[n] = -__expf(logA[d * kNState + n]);
  const float Dd = Dp[d];

  float h[kNState];
#pragma unroll
  for (int n = 0; n < kNState; ++n) h[n] = 0.f;

  for (int t = 0; t < kT; ++t) {
    const size_t m  = (size_t)b * kT + t;
    const float dtv = bf2f(dtb[m * kDn + d]);
    const float uv  = bf2f(ub[m * kDn + d]);
    const float* bc = xbc + m * kXbcW;            // [64..79]=B_t, [80..95]=C_t
    const float zv  = bf2f(xzb[m * (2 * kDn) + kDn + d]);
    const float du  = dtv * uv;
    float y = 0.f;
#pragma unroll
    for (int n = 0; n < kNState; ++n) {
      h[n] = __expf(dtv * Ad[n]) * h[n] + du * bc[64 + n];
      y += h[n] * bc[80 + n];
    }
    const float sz = zv / (1.f + __expf(-zv));    // silu(z)
    yb[m * kDn + d] = f2bf((y + uv * Dd) * sz);
  }
}

// ---------------- host side ----------------
extern "C" void kernel_launch(void* const* d_in, const int* in_sizes, int n_in,
                              void* d_out, int out_size, void* d_ws, size_t ws_size,
                              hipStream_t stream) {
  (void)in_sizes; (void)n_in; (void)out_size; (void)ws_size;
  const float* x      = (const float*)d_in[0];   // (B,T,Dm)
  const float* W_in   = (const float*)d_in[1];   // (2*Dn, Dm)
  const float* conv_w = (const float*)d_in[2];   // (Dn,1,4)
  const float* conv_b = (const float*)d_in[3];   // (Dn)
  const float* W_x    = (const float*)d_in[4];   // (96, Dn)
  const float* W_dt   = (const float*)d_in[5];   // (Dn, 64)
  const float* b_dt   = (const float*)d_in[6];   // (Dn)
  const float* log_A  = (const float*)d_in[7];   // (Dn, 16)
  const float* D_par  = (const float*)d_in[8];   // (Dn)
  const float* W_out  = (const float*)d_in[9];   // (Dm, Dn)
  const float* ln_w   = (const float*)d_in[10];  // (Dm)
  const float* ln_b   = (const float*)d_in[11];  // (Dm)
  float* out = (float*)d_out;                    // (B,T,Dm) f32

  char* ws = (char*)d_ws;
  size_t off = 0;
  auto carve = [&](size_t bytes) -> char* {
    char* p = ws + off; off += (bytes + 255) & ~(size_t)255; return p;
  };
  unsigned short* xn_bf   = (unsigned short*)carve((size_t)kM * kDm * 2);
  unsigned short* Win_bf  = (unsigned short*)carve((size_t)2 * kDn * kDm * 2);
  unsigned short* Wx_bf   = (unsigned short*)carve((size_t)kXbcW * kDn * 2);
  unsigned short* Wdt_bf  = (unsigned short*)carve((size_t)kDn * kDtRank * 2);
  unsigned short* Wout_bf = (unsigned short*)carve((size_t)kDm * kDn * 2);
  unsigned short* xz_bf   = (unsigned short*)carve((size_t)kM * 2 * kDn * 2);
  unsigned short* u_bf    = (unsigned short*)carve((size_t)kM * kDn * 2);
  float*          xbc_f   = (float*)carve((size_t)kM * kXbcW * 4);
  unsigned short* dtl_bf  = (unsigned short*)carve((size_t)kM * kDtRank * 2);
  unsigned short* dt_bf   = (unsigned short*)carve((size_t)kM * kDn * 2);
  unsigned short* y_bf    = (unsigned short*)carve((size_t)kM * kDn * 2);

  // 1. weight conversions to bf16
  {
    int n;
    n = 2 * kDn * kDm;   cvt_bf16_kernel<<<(n + 255) / 256, 256, 0, stream>>>(W_in,  Win_bf,  n);
    n = kXbcW * kDn;     cvt_bf16_kernel<<<(n + 255) / 256, 256, 0, stream>>>(W_x,   Wx_bf,   n);
    n = kDn * kDtRank;   cvt_bf16_kernel<<<(n + 255) / 256, 256, 0, stream>>>(W_dt,  Wdt_bf,  n);
    n = kDm * kDn;       cvt_bf16_kernel<<<(n + 255) / 256, 256, 0, stream>>>(W_out, Wout_bf, n);
  }

  // 2. LayerNorm -> xn (bf16)
  ln_kernel<<<kM, 256, 0, stream>>>(x, ln_w, ln_b, xn_bf);

  // 3. xz = xn @ W_in^T   (M=16384, N=4096, K=1024), TDM-staged A, -> bf16
  gemm_lds_wmma<0, 128><<<(kM / 16) * (2 * kDn / 512), 256, 0, stream>>>(
      (const bf16_t*)xn_bf, (const bf16_t*)Win_bf, kM, 2 * kDn, kDm,
      nullptr, nullptr, xz_bf);

  // 4. causal depthwise conv + SiLU -> u (bf16)
  conv_silu_kernel<<<(kM * kDn + 255) / 256, 256, 0, stream>>>(xz_bf, conv_w, conv_b, u_bf);

  // 5. xbc = u @ W_x^T    (M=16384, N=96, K=2048) -> f32
  gemm_plain_wmma<1, 6><<<(kM / 16) / 8, 256, 0, stream>>>(
      (const bf16_t*)u_bf, (const bf16_t*)Wx_bf, kM, kXbcW, kDn,
      nullptr, xbc_f, nullptr);

  // 6. dt_low slice -> bf16
  slice_dtlow_kernel<<<(kM * kDtRank + 255) / 256, 256, 0, stream>>>(xbc_f, dtl_bf);

  // 7. dt = clip(softplus(dt_low @ W_dt^T + b_dt)+1e-3)  (N=2048, K=64) -> bf16
  gemm_lds_wmma<2, 64><<<(kM / 16) * (kDn / 512), 256, 0, stream>>>(
      (const bf16_t*)dtl_bf, (const bf16_t*)Wdt_bf, kM, kDn, kDtRank,
      b_dt, nullptr, dt_bf);

  // 8. selective scan + D skip + SiLU(z) gate -> y (bf16)
  scan_kernel<<<(kB * kDn + 255) / 256, 256, 0, stream>>>(
      dt_bf, u_bf, xbc_f, xz_bf, log_A, D_par, y_bf);

  // 9. out = y @ W_out^T + residual   (N=1024, K=2048) -> f32
  gemm_lds_wmma<3, 128><<<(kM / 16) * (kDm / 512), 256, 0, stream>>>(
      (const bf16_t*)y_bf, (const bf16_t*)Wout_bf, kM, kDm, kDn,
      x, out, nullptr);
}
